// MultiObjectiveCriticNetwork_10033043603499
// MI455X (gfx1250) — matrix operand order repaired and checked by reading, not compiled, not run on hardware
//
#include <hip/hip_runtime.h>
#include <hip/hip_bf16.h>

typedef __attribute__((ext_vector_type(16))) _Float16 v16h;
typedef __attribute__((ext_vector_type(8)))  _Float16 v8h;
typedef __attribute__((ext_vector_type(8)))  float    v8f;
typedef __attribute__((ext_vector_type(4)))  unsigned int v4u;
typedef __attribute__((ext_vector_type(8)))  int      v8i;
typedef __attribute__((ext_vector_type(4)))  int      v4i;

#define WAVES_PER_WG 4
#define ROWS_PER_WG  (WAVES_PER_WG * 16)
#define OBS_DIM      512
#define START_IDX    68
#define NSRV         64
#define OBS_WO       448   // 512 - 64
#define COMB         640   // 448 + 128 + 64

// ws layout (in halves), fragment-packed f16 weights
#define W1_OFF  0            // h_w1  Kpad=32  N=128 : 1x8   frags (direct)
#define W2_OFF  4096         // h_w2  K=128    N=128 : 4x8   frags (direct)
#define W3_OFF  20480        // s_w1  K=640    N=256 : 20x16 frags (TDM staged)
#define W4_OFF  184320       // s_w2  K=256    N=256 : 8x16  frags (TDM staged)
#define W56_OFF 249856       // qd|qe K=256    N=64+64: 8x(4+4) frags (TDM staged)
#define WS_HALVES 282624

// ---------------------------------------------------------------------------
// Wave-local LDS fence (LDS is in-order per wave; this pins compiler ordering
// and drains DScnt at cross-lane producer/consumer points).
__device__ __forceinline__ void wave_lds_sync() {
    __builtin_amdgcn_wave_barrier();
    asm volatile("s_wait_dscnt 0" ::: "memory");
    __builtin_amdgcn_wave_barrier();
}

__device__ __forceinline__ v8f wmma16(v16h a, v16h b, v8f c) {
    // D = A(16x32 f16) * B(32x16 f16) + C(16x16 f32)
    return __builtin_amdgcn_wmma_f32_16x16x32_f16(
        false, a, false, b, (short)0, c, false, false);
}

// A-fragment from LDS, row-major [16][strideH] f16 tile, k-step kt (K=32).
// ISA layout: lanes 0-15 hold M=lane, K {0..7,16..23}; lanes 16-31 K {8..15,24..31}.
__device__ __forceinline__ v16h a_frag(const _Float16* base, int strideH, int kt, int lane) {
    const int m = lane & 15, hsel = lane >> 4;
    const _Float16* p = base + m * strideH + kt * 32 + hsel * 8;
    v8h lo = *(const v8h*)(p);        // K = k0 .. k0+7
    v8h hi = *(const v8h*)(p + 16);   // K = k0+16 .. k0+23
    return __builtin_shufflevector(lo, hi, 0,1,2,3,4,5,6,7,8,9,10,11,12,13,14,15);
}

// B-fragment: pre-swizzled 512-half blob, lane-major 16 halves (2 x b128).
__device__ __forceinline__ v16h b_frag(const _Float16* frag, int lane) {
    const _Float16* p = frag + lane * 16;
    v8h lo = *(const v8h*)(p);
    v8h hi = *(const v8h*)(p + 8);
    return __builtin_shufflevector(lo, hi, 0,1,2,3,4,5,6,7,8,9,10,11,12,13,14,15);
}

// ---------------------------------------------------------------------------
// Tensor Data Mover: 1D contiguous copy global -> LDS, data_size=8B units.
// D# packing per CDNA5 ISA:
//   g0: count=1 | lds_addr | global_addr[56:0] | type=2
//   g1: data_size=3(8B), tensor_dim0=n8, tensor_dim1=1, tile_dim0=n8,
//       tile_dim1=1, tensor_dim0_stride=n8
// Groups 2/3 unused (<=2D tensor) -> zero-filled.
__device__ __forceinline__ void tdm_copy(const _Float16* gsrc, void* ldst, unsigned bytes) {
    const unsigned long long ga = (unsigned long long)(size_t)gsrc;
    const unsigned ldsOff = (unsigned)(size_t)ldst;   // flat low 32 bits = LDS byte addr
    const unsigned n8 = bytes >> 3;                   // 8-byte elements
    v4u g0;
    g0.x = 1u;                                        // count = 1 (valid descriptor)
    g0.y = ldsOff;                                    // lds_addr (bits 63:32)
    g0.z = (unsigned)ga;                              // global_addr[31:0]
    g0.w = (unsigned)((ga >> 32) & 0x01FFFFFFu) | (2u << 30);  // ga[56:32] | type=2
    v8i g1;
    g1[0] = (int)(3u << 16);                          // workgroup_mask=0, data_size=3 (8B)
    g1[1] = (int)((n8 & 0xFFFFu) << 16);              // tensor_dim0[15:0] @ bits 63:48
    g1[2] = (int)(((n8 >> 16) & 0xFFFFu) | (1u << 16)); // tensor_dim0[31:16] | tensor_dim1=1
    g1[3] = (int)((n8 & 0xFFFFu) << 16);              // tile_dim0 @ bits 127:112
    g1[4] = 1;                                        // tile_dim1 = 1, tile_dim2 = 0
    g1[5] = (int)n8;                                  // tensor_dim0_stride[31:0]
    g1[6] = 0;
    g1[7] = 0;
    v4i z4 = {0, 0, 0, 0};
    v8i z8 = {0, 0, 0, 0, 0, 0, 0, 0};
    __builtin_amdgcn_tensor_load_to_lds(g0, g1, z4, z4, z8, 0);
}

// ---------------------------------------------------------------------------
// GEMM with B streamed from global blob directly (small h-layers).
__device__ __forceinline__ void gemm_bias_relu_to_lds(
    const _Float16* aBase, int aStride, int ktiles,
    const _Float16* blob, int ntiles,
    _Float16* dBase, int dStride, int dColOff,
    const float* bias, int lane)
{
    const int nl = lane & 15, mh = (lane >> 4) * 8;
    for (int nc = 0; nc < ntiles; nc += 4) {
        v8f acc[4] = {};
        for (int kt = 0; kt < ktiles; ++kt) {
            v16h a = a_frag(aBase, aStride, kt, lane);
            const _Float16* bp = blob + ((size_t)kt * ntiles + nc) * 512;
#pragma unroll
            for (int t = 0; t < 4; ++t)
                acc[t] = wmma16(a, b_frag(bp + t * 512, lane), acc[t]);
        }
#pragma unroll
        for (int t = 0; t < 4; ++t) {
            const int n = (nc + t) * 16 + nl;
            const float bn = bias[n];
#pragma unroll
            for (int r = 0; r < 8; ++r) {
                float v = acc[t][r] + bn;
                v = v > 0.f ? v : 0.f;
                dBase[(mh + r) * dStride + dColOff + n] = (_Float16)v;
            }
        }
    }
}

// GEMM with B-slices TDM-staged into workgroup-shared LDS, double buffered.
// One kt-slice = NT fragments = NT*1024 bytes. Wave 0 drives the TDM; the
// whole WG consumes the shared copy (weight traffic / WAVES_PER_WG).
template<int NT>
__device__ __forceinline__ void gemm_staged(
    const _Float16* aBase, int aStride, int ktiles,
    const _Float16* blob, _Float16* wb0, _Float16* wb1,
    v8f* acc, int lane, int wid)
{
    const unsigned sliceBytes = NT * 512u * 2u;
    if (wid == 0) {
        tdm_copy(blob, wb0, sliceBytes);
        __builtin_amdgcn_s_wait_tensorcnt(0);
    }
    __syncthreads();
    for (int kt = 0; kt < ktiles; ++kt) {
        _Float16* cur = (kt & 1) ? wb1 : wb0;
        _Float16* nxt = (kt & 1) ? wb0 : wb1;
        const bool more = (kt + 1) < ktiles;
        if (wid == 0 && more)   // prefetch next slice while computing this one
            tdm_copy(blob + (size_t)(kt + 1) * NT * 512, nxt, sliceBytes);
        v16h a = a_frag(aBase, aStride, kt, lane);
#pragma unroll
        for (int t = 0; t < NT; ++t)
            acc[t] = wmma16(a, b_frag(cur + t * 512, lane), acc[t]);
        if (wid == 0 && more)
            __builtin_amdgcn_s_wait_tensorcnt(0);
        __syncthreads();        // publish next slice / protect cur from reuse
    }
}

// In-place LayerNorm over f16 rows in LDS (wave32: halves combined via shfl_xor 16).
__device__ __forceinline__ void ln_rows(
    _Float16* base, int stride, int colOff, int nFeat,
    const float* g, const float* b, int lane)
{
    const int row = lane & 15, half = lane >> 4;
    const int h = nFeat >> 1, i0 = half * h;
    _Float16* p = base + row * stride + colOff;
    float sum = 0.f, sq = 0.f;
    for (int i = i0; i < i0 + h; ++i) {
        float x = (float)p[i];
        sum += x; sq += x * x;
    }
    sum += __shfl_xor(sum, 16);
    sq  += __shfl_xor(sq, 16);
    const float mu = sum / (float)nFeat;
    const float var = sq / (float)nFeat - mu * mu;
    const float rinv = rsqrtf(var + 1e-5f);
    for (int i = i0; i < i0 + h; ++i) {
        float x = (float)p[i];
        p[i] = (_Float16)((x - mu) * rinv * g[i] + b[i]);
    }
}

// ---------------------------------------------------------------------------
// Pack f32 weight [K][N] -> fragment-major f16 blob (zero-pad K to 32 multiple).
// Fragment (kt, nt) lands at ((kt*ntStride + ntBase + nt) * 512); within a
// fragment, element (lane, i): k = kt*32 + (lane>>4)*8 + (i&7) + (i>>3)*16, col = nt*16 + (lane&15).
__global__ void pack_weights_kernel(const float* __restrict__ src,
                                    _Float16* __restrict__ dst,
                                    int K, int N, int ntiles,
                                    int ntStride, int ntBase, int total)
{
    int tid = blockIdx.x * blockDim.x + threadIdx.x;
    if (tid >= total) return;
    const int frag = tid >> 9;
    const int rem  = tid & 511;
    const int lane = rem >> 4;
    const int i    = rem & 15;
    const int kt   = frag / ntiles;
    const int nt   = frag - kt * ntiles;
    const int k    = kt * 32 + (lane >> 4) * 8 + (i & 7) + (i >> 3) * 16;
    const int c    = nt * 16 + (lane & 15);
    float v = (k < K) ? src[(size_t)k * N + c] : 0.f;
    dst[(((size_t)kt * ntStride + ntBase + nt) << 9) | rem] = (_Float16)v;
}

// ---------------------------------------------------------------------------
__global__ __launch_bounds__(WAVES_PER_WG * 32, 1)
void critic_main_kernel(
    const float* __restrict__ obs, const float* __restrict__ pref,
    const float* __restrict__ h_b1, const float* __restrict__ h1g, const float* __restrict__ h1b,
    const float* __restrict__ h_b2, const float* __restrict__ h2g, const float* __restrict__ h2b,
    const float* __restrict__ p_w,  const float* __restrict__ p_b,
    const float* __restrict__ plg,  const float* __restrict__ plb,
    const float* __restrict__ s_b1, const float* __restrict__ s1g, const float* __restrict__ s1b,
    const float* __restrict__ s_b2, const float* __restrict__ s2g, const float* __restrict__ s2b,
    const float* __restrict__ qd_b, const float* __restrict__ qe_b,
    const _Float16* __restrict__ ws, float* __restrict__ out)
{
    __shared__ __align__(16) _Float16 s_aT[WAVES_PER_WG][16][COMB]; // combined A tile
    __shared__ __align__(16) _Float16 s_st[WAVES_PER_WG][16][256];  // stage buffer
    __shared__ __align__(16) _Float16 s_hh[WAVES_PER_WG][16][32];   // hist A tile
    __shared__ __align__(16) _Float16 s_wb[2][16 * 512];            // TDM weight slices

    const int lane = threadIdx.x & 31;
    const int wid  = threadIdx.x >> 5;
    const int rowBase = (blockIdx.x * WAVES_PER_WG + wid) * 16;
    _Float16* aT = &s_aT[wid][0][0];
    _Float16* st = &s_st[wid][0][0];
    _Float16* hh = &s_hh[wid][0][0];
    const int row = lane & 15, half = lane >> 4;

    // ---- zero padded hist tile (covers K=10..31) ----
    for (int i = lane; i < 16 * 32; i += 32) hh[i] = (_Float16)0.f;

    // ---- per-row histogram: lane pair (row, half) each bins 32 workloads ----
    float bins[10];
#pragma unroll
    for (int b = 0; b < 10; ++b) bins[b] = 0.f;
    {
        const float* wl = obs + (size_t)(rowBase + row) * OBS_DIM + START_IDX + half * 32;
        for (int j = 0; j < 32; ++j) {
            float x = wl[j];
            float inc = (x >= 0.f && x <= 10.f) ? 1.f : 0.f;
            int bi = (int)floorf(x);
            bi = bi < 0 ? 0 : (bi > 9 ? 9 : bi);
#pragma unroll
            for (int b = 0; b < 10; ++b) bins[b] += (bi == b) ? inc : 0.f;
        }
    }
    float tot = 0.f;
#pragma unroll
    for (int b = 0; b < 10; ++b) {
        bins[b] += __shfl_xor(bins[b], 16);
        tot += bins[b];
    }
    const float hinv = 1.f / (tot + 1e-8f);
    wave_lds_sync();
#pragma unroll
    for (int b = 0; b < 10; ++b)   // lane L and L+16 write identical values
        hh[row * 32 + b] = (_Float16)(bins[b] * hinv);
    wave_lds_sync();

    // ---- hist MLP: 10->128 (K padded to 32), LN ----
    gemm_bias_relu_to_lds(hh, 32, 1, ws + W1_OFF, 8, st, 256, 0, h_b1, lane);
    wave_lds_sync();
    ln_rows(st, 256, 0, 128, h1g, h1b, lane);
    wave_lds_sync();
    // ---- 128->128, result lands in combined tile cols [448,576), LN in place ----
    gemm_bias_relu_to_lds(st, 256, 4, ws + W2_OFF, 8, aT, COMB, OBS_WO, h_b2, lane);
    wave_lds_sync();
    ln_rows(aT, COMB, OBS_WO, 128, h2g, h2b, lane);

    // ---- preference MLP 2->64 (scalar, two-pass LN), cols [576,640) ----
    {
        const float p0 = pref[(size_t)(rowBase + row) * 2 + 0];
        const float p1 = pref[(size_t)(rowBase + row) * 2 + 1];
        float sum = 0.f, sq = 0.f;
        for (int n = 0; n < 64; ++n) {
            float v = p0 * p_w[n] + p1 * p_w[64 + n] + p_b[n];
            v = v > 0.f ? v : 0.f;
            sum += v; sq += v * v;
        }
        const float mu = sum * (1.f / 64.f);
        const float var = sq * (1.f / 64.f) - mu * mu;
        const float rinv = rsqrtf(var + 1e-5f);
        for (int n = 0; n < 64; ++n) {
            float v = p0 * p_w[n] + p1 * p_w[64 + n] + p_b[n];
            v = v > 0.f ? v : 0.f;
            aT[row * COMB + OBS_WO + 128 + n] = (_Float16)((v - mu) * rinv * plg[n] + plb[n]);
        }
    }

    // ---- obs_wo: copy 448 kept features (skip the 64 workload cols), cvt f16 ----
    for (int r = 0; r < 16; ++r) {
        const float* src = obs + (size_t)(rowBase + r) * OBS_DIM;
#pragma unroll 2
        for (int j = 0; j < 14; ++j) {
            const int c = j * 32 + lane;
            const int sc = c < START_IDX ? c : c + NSRV;
            aT[r * COMB + c] = (_Float16)src[sc];
        }
    }
    wave_lds_sync();

    // ---- trunk 640->256 with TDM-staged weights, LN ----
    {
        v8f acc[16] = {};
        gemm_staged<16>(aT, COMB, 20, ws + W3_OFF, &s_wb[0][0], &s_wb[1][0],
                        acc, lane, wid);
        const int nl = lane & 15, mh = (lane >> 4) * 8;
#pragma unroll
        for (int t = 0; t < 16; ++t) {
            const int n = t * 16 + nl;
            const float bn = s_b1[n];
#pragma unroll
            for (int r = 0; r < 8; ++r) {
                float v = acc[t][r] + bn;
                v = v > 0.f ? v : 0.f;
                st[(mh + r) * 256 + n] = (_Float16)v;
            }
        }
    }
    wave_lds_sync();
    ln_rows(st, 256, 0, 256, s1g, s1b, lane);
    wave_lds_sync();

    // ---- trunk 256->256 with TDM-staged weights, LN ----
    {
        v8f acc[16] = {};
        gemm_staged<16>(st, 256, 8, ws + W4_OFF, &s_wb[0][0], &s_wb[1][0],
                        acc, lane, wid);
        const int nl = lane & 15, mh = (lane >> 4) * 8;
#pragma unroll
        for (int t = 0; t < 16; ++t) {
            const int n = t * 16 + nl;
            const float bn = s_b2[n];
#pragma unroll
            for (int r = 0; r < 8; ++r) {
                float v = acc[t][r] + bn;
                v = v > 0.f ? v : 0.f;
                aT[(mh + r) * COMB + n] = (_Float16)v;
            }
        }
    }
    wave_lds_sync();
    ln_rows(aT, COMB, 0, 256, s2g, s2b, lane);
    wave_lds_sync();

    // ---- dual heads 256->(64|64), interleaved blob, TDM staged ----
    {
        v8f acc[8] = {};
        gemm_staged<8>(aT, COMB, 8, ws + W56_OFF, &s_wb[0][0], &s_wb[1][0],
                       acc, lane, wid);
        const int nl = lane & 15, mh = (lane >> 4) * 8;
#pragma unroll
        for (int t = 0; t < 4; ++t) {
            const int n = t * 16 + nl;
            const float bd = qd_b[n], be = qe_b[n];
#pragma unroll
            for (int r = 0; r < 8; ++r) {
                const int m = mh + r;
                const size_t o = ((size_t)(rowBase + m) * 64 + n) * 2;
                float2 v2;
                v2.x = acc[t][r] + bd;      // q_delay
                v2.y = acc[t + 4][r] + be;  // q_energy
                *(float2*)(out + o) = v2;
            }
        }
    }
}

// ---------------------------------------------------------------------------
extern "C" void kernel_launch(void* const* d_in, const int* in_sizes, int n_in,
                              void* d_out, int out_size, void* d_ws, size_t ws_size,
                              hipStream_t stream) {
    const float* obs  = (const float*)d_in[0];
    const float* pref = (const float*)d_in[1];
    const float* h_w1 = (const float*)d_in[2];
    const float* h_b1 = (const float*)d_in[3];
    const float* h1g  = (const float*)d_in[4];
    const float* h1b  = (const float*)d_in[5];
    const float* h_w2 = (const float*)d_in[6];
    const float* h_b2 = (const float*)d_in[7];
    const float* h2g  = (const float*)d_in[8];
    const float* h2b  = (const float*)d_in[9];
    const float* p_w  = (const float*)d_in[10];
    const float* p_b  = (const float*)d_in[11];
    const float* plg  = (const float*)d_in[12];
    const float* plb  = (const float*)d_in[13];
    const float* s_w1 = (const float*)d_in[14];
    const float* s_b1 = (const float*)d_in[15];
    const float* s1g  = (const float*)d_in[16];
    const float* s1b  = (const float*)d_in[17];
    const float* s_w2 = (const float*)d_in[18];
    const float* s_b2 = (const float*)d_in[19];
    const float* s2g  = (const float*)d_in[20];
    const float* s2b  = (const float*)d_in[21];
    const float* qd_w = (const float*)d_in[22];
    const float* qd_b = (const float*)d_in[23];
    const float* qe_w = (const float*)d_in[24];
    const float* qe_b = (const float*)d_in[25];

    _Float16* ws = (_Float16*)d_ws;
    float* out = (float*)d_out;

    const int B = in_sizes[0] / OBS_DIM;   // 65536

    auto pk = [&](const float* src, int off, int K, int N, int nt,
                  int ntStride, int ntBase, int total) {
        pack_weights_kernel<<<(total + 255) / 256, 256, 0, stream>>>(
            src, ws + off, K, N, nt, ntStride, ntBase, total);
    };
    pk(h_w1, W1_OFF, 10, 128, 8, 8, 0, 4096);
    pk(h_w2, W2_OFF, 128, 128, 8, 8, 0, 16384);
    pk(s_w1, W3_OFF, 640, 256, 16, 16, 0, 163840);
    pk(s_w2, W4_OFF, 256, 256, 16, 16, 0, 65536);
    pk(qd_w, W56_OFF, 256, 64, 4, 8, 0, 16384);   // interleaved head blob: [qd|qe] per kt
    pk(qe_w, W56_OFF, 256, 64, 4, 8, 4, 16384);

    const int grid = B / ROWS_PER_WG;
    critic_main_kernel<<<grid, WAVES_PER_WG * 32, 0, stream>>>(
        obs, pref,
        h_b1, h1g, h1b,
        h_b2, h2g, h2b,
        p_w, p_b, plg, plb,
        s_b1, s1g, s1b,
        s_b2, s2g, s2b,
        qd_b, qe_b,
        ws, out);
}
